// BasicDeformableConv2D_44375602102286
// MI455X (gfx1250) — compile-verified, matrix operand order checked by reading, not run on hardware
//
#include <hip/hip_runtime.h>

typedef __attribute__((ext_vector_type(16))) _Float16 v16h;
typedef __attribute__((ext_vector_type(8)))  _Float16 v8h;
typedef __attribute__((ext_vector_type(8)))  float    v8f;

// Problem constants (fixed by the reference)
#define NB   4
#define CIN  256
#define HH   64
#define WW   64
#define COUT 256
#define KT   9          // 3x3 taps
#define LPIX 4096       // Ho*Wo
#define KDIM 2304       // CIN*KT  (GEMM K)

// ---------------------------------------------------------------------------
// CDNA5 async global->LDS copy (ASYNCcnt-tracked, no VGPR round trip)
// ---------------------------------------------------------------------------
__device__ __forceinline__ unsigned lds_addr32(const void* p) {
    // flat LDS addresses: LDS address is the pointer's low 32 bits (ISA 10.2 aperture rules)
    return (unsigned)(uintptr_t)p;
}

__device__ __forceinline__ void async_ld_b128(unsigned lds, const void* g) {
    asm volatile("global_load_async_to_lds_b128 %0, %1, off"
                 :: "v"(lds), "v"(g) : "memory");
}

__device__ __forceinline__ void wait_async_le4() {
    asm volatile("s_wait_asynccnt 0x4" ::: "memory");
}
__device__ __forceinline__ void wait_async_0() {
    asm volatile("s_wait_asynccnt 0x0" ::: "memory");
}

// ---------------------------------------------------------------------------
// Kernel 1: filter f32 -> f16, layout [Co][C*9] (K index = c*9 + k)
// ---------------------------------------------------------------------------
__global__ __launch_bounds__(256) void k_wconv(const float* __restrict__ wf,
                                               _Float16* __restrict__ wa, int n) {
    int i = blockIdx.x * 256 + threadIdx.x;
    if (i < n) wa[i] = (_Float16)wf[i];
}

// ---------------------------------------------------------------------------
// Kernel 2: deformable im2col -> f16 matrix  B[n][l][c*9+k]  (K-contiguous rows)
// grid = (L/64, N), block = 256
// ---------------------------------------------------------------------------
__global__ __launch_bounds__(256) void k_im2col(const float* __restrict__ x,
                                                const float* __restrict__ off,
                                                const float* __restrict__ msk,
                                                _Float16* __restrict__ Bm) {
    const int n  = blockIdx.y;
    const int l0 = blockIdx.x * 64;

    __shared__ float sw[64 * KT][4];   // bilinear weights * mask * validity
    __shared__ int   si[64 * KT][4];   // clamped corner indices into HxW plane

    const int t = threadIdx.x;

    // Phase A: per-(pixel, tap) geometry -> LDS
    for (int item = t; item < 64 * KT; item += 256) {
        const int ll = item / KT;
        const int k  = item - ll * KT;
        const int l  = l0 + ll;
        const int ho = l >> 6;
        const int wo = l & 63;

        const float by = (float)((k / 3) + ho - 1);   // dil=1, pad=1, stride=1
        const float bx = (float)((k % 3) + wo - 1);

        const int ob = ((n * (2 * KT) + k * 2) * HH + ho) * WW + wo;
        const float py = by + off[ob];                 // y offset channel (k*2+0)
        const float px = bx + off[ob + HH * WW];       // x offset channel (k*2+1)
        const float m  = msk[((n * KT + k) * HH + ho) * WW + wo];

        const float y0f = floorf(py);
        const float x0f = floorf(px);
        const float ly = py - y0f;
        const float lx = px - x0f;
        const int y0 = (int)y0f, x0 = (int)x0f;
        const int y1 = y0 + 1,  x1 = x0 + 1;

        const float vy0 = (y0 >= 0 && y0 < HH) ? 1.f : 0.f;
        const float vy1 = (y1 >= 0 && y1 < HH) ? 1.f : 0.f;
        const float vx0 = (x0 >= 0 && x0 < WW) ? 1.f : 0.f;
        const float vx1 = (x1 >= 0 && x1 < WW) ? 1.f : 0.f;

        const int cy0 = min(max(y0, 0), HH - 1);
        const int cy1 = min(max(y1, 0), HH - 1);
        const int cx0 = min(max(x0, 0), WW - 1);
        const int cx1 = min(max(x1, 0), WW - 1);

        sw[item][0] = (1.f - ly) * (1.f - lx) * m * vy0 * vx0;
        sw[item][1] = (1.f - ly) * lx         * m * vy0 * vx1;
        sw[item][2] = ly         * (1.f - lx) * m * vy1 * vx0;
        sw[item][3] = ly         * lx         * m * vy1 * vx1;
        si[item][0] = cy0 * WW + cx0;
        si[item][1] = cy0 * WW + cx1;
        si[item][2] = cy1 * WW + cx0;
        si[item][3] = cy1 * WW + cx1;
    }
    __syncthreads();

    // Phase B: stream all channels, write f16 rows (l-major, K-contiguous)
    const int ll = t >> 2;         // 64 pixels, 4 threads each
    const int c0 = t & 3;
    const int l  = l0 + ll;
    const size_t rowbase = ((size_t)n * LPIX + l) * KDIM;

    for (int c = c0; c < CIN; c += 4) {
        const float* __restrict__ xc = x + (((size_t)n * CIN + c) << 12); // *4096
        _Float16* __restrict__ dst = Bm + rowbase + (size_t)c * KT;
#pragma unroll
        for (int k = 0; k < KT; ++k) {
            const float* w = sw[ll * KT + k];
            const int*   ii = si[ll * KT + k];
            float s = w[0] * xc[ii[0]] + w[1] * xc[ii[1]] +
                      w[2] * xc[ii[2]] + w[3] * xc[ii[3]];
            dst[k] = (_Float16)s;
        }
    }
}

// ---------------------------------------------------------------------------
// Kernel 3: WMMA GEMM  out[n][co][l] = sum_kc A[co][kc] * B[n][l][kc]
// block = 256 (8 waves), tile = 128(M) x 128(N), K step = 32,
// double-buffered LDS filled by async global->LDS copies (software pipeline).
// grid = (L/128, Co/128, N)
// ---------------------------------------------------------------------------
__device__ __forceinline__ v16h ld_frag(const _Float16* p) {
    // per-lane fragment: K = {k0..k0+7} and {k0+16..k0+23} relative to tile row
    v8h lo = *(const v8h*)(p);
    v8h hi = *(const v8h*)(p + 16);
    v16h r;
#pragma unroll
    for (int i = 0; i < 8; ++i) { r[i] = lo[i]; r[i + 8] = hi[i]; }
    return r;
}

__global__ __launch_bounds__(256) void k_gemm(const _Float16* __restrict__ A,
                                              const _Float16* __restrict__ B,
                                              float* __restrict__ out) {
    const int n  = blockIdx.z;
    const int m0 = blockIdx.y * 128;
    const int l0 = blockIdx.x * 128;

    __shared__ _Float16 As[2][128 * 32];   // [buf][row][k]  2 x 8 KB
    __shared__ _Float16 Bs[2][128 * 32];   // [buf][col][k]  2 x 8 KB

    const int t    = threadIdx.x;
    const int lane = t & 31;
    const int w    = t >> 5;
    const int wm   = (w & 1) * 64;      // wave M offset within block tile
    const int wn   = (w >> 1) * 32;     // wave N offset within block tile
    const int lm   = lane & 15;
    const int k0   = (lane >> 4) * 8;   // 16-bit A/B lane K-group
    const int mhi  = (lane >> 4) * 8;   // C/D: lanes 16-31 hold M+8

    // per-thread staging slots: chunks t and t+256 of 512 (row = id/4, k8 = (id%4)*8)
    const int r0 = t >> 2;              // rows 0..63
    const int r1 = r0 + 64;             // rows 64..127
    const int k8 = (t & 3) * 8;

    const _Float16* __restrict__ Ag = A + (size_t)m0 * KDIM;
    const _Float16* __restrict__ Bg = B + ((size_t)n * LPIX + l0) * KDIM;

    v8f acc[4][2];
#pragma unroll
    for (int i = 0; i < 4; ++i)
#pragma unroll
        for (int j = 0; j < 2; ++j)
#pragma unroll
            for (int v = 0; v < 8; ++v) acc[i][j][v] = 0.f;

    // issue one 128x32 A slab + 128x32 B slab into buffer `buf` (4 async b128/thread)
    auto issue = [&](int buf, int kt) {
        async_ld_b128(lds_addr32(&As[buf][r0 * 32 + k8]), &Ag[(size_t)r0 * KDIM + kt + k8]);
        async_ld_b128(lds_addr32(&As[buf][r1 * 32 + k8]), &Ag[(size_t)r1 * KDIM + kt + k8]);
        async_ld_b128(lds_addr32(&Bs[buf][r0 * 32 + k8]), &Bg[(size_t)r0 * KDIM + kt + k8]);
        async_ld_b128(lds_addr32(&Bs[buf][r1 * 32 + k8]), &Bg[(size_t)r1 * KDIM + kt + k8]);
    };

    issue(0, 0);                         // prologue: fill buffer 0

    for (int kt = 0; kt < KDIM; kt += 32) {
        const int cur = (kt >> 5) & 1;
        if (kt + 32 < KDIM) {
            issue(cur ^ 1, kt + 32);     // overlap next slab with this slab's WMMAs
            wait_async_le4();            // in-order completion => current slab landed
        } else {
            wait_async_0();
        }
        __syncthreads();                 // all waves' slabs visible in LDS

        v16h a[4], b[2];
#pragma unroll
        for (int i = 0; i < 4; ++i)
            a[i] = ld_frag(&As[cur][(wm + i * 16 + lm) * 32 + k0]);
#pragma unroll
        for (int j = 0; j < 2; ++j)
            b[j] = ld_frag(&Bs[cur][(wn + j * 16 + lm) * 32 + k0]);

#pragma unroll
        for (int i = 0; i < 4; ++i)
#pragma unroll
            for (int j = 0; j < 2; ++j)
                acc[i][j] = __builtin_amdgcn_wmma_f32_16x16x32_f16(
                    false, a[i], false, b[j], (short)0, acc[i][j], false, false);

        __syncthreads();                 // reads done before buffer is overwritten
    }

    // write out: VGPR v <-> M = v + mhi, N = lane&15 (16x16 f32 C/D layout)
#pragma unroll
    for (int i = 0; i < 4; ++i)
#pragma unroll
        for (int j = 0; j < 2; ++j) {
            const int co = m0 + wm + i * 16 + mhi;
            const int l  = l0 + wn + j * 16 + lm;
            float* __restrict__ o = out + (((size_t)n * COUT + co) << 12) + l;
#pragma unroll
            for (int v = 0; v < 8; ++v) o[(size_t)v << 12] = acc[i][j][v];
        }
}

// ---------------------------------------------------------------------------
extern "C" void kernel_launch(void* const* d_in, const int* in_sizes, int n_in,
                              void* d_out, int out_size, void* d_ws, size_t ws_size,
                              hipStream_t stream) {
    const float* x   = (const float*)d_in[0];
    const float* wf  = (const float*)d_in[1];
    const float* off = (const float*)d_in[2];
    const float* msk = (const float*)d_in[3];
    float* out = (float*)d_out;

    // workspace layout: [ B f16: N*L*KDIM ][ A f16: COUT*KDIM ]
    _Float16* Bws = (_Float16*)d_ws;
    _Float16* Aws = (_Float16*)((char*)d_ws + (size_t)NB * LPIX * KDIM * sizeof(_Float16));

    const int wn = COUT * KDIM;                         // 589824
    k_wconv<<<(wn + 255) / 256, 256, 0, stream>>>(wf, Aws, wn);
    k_im2col<<<dim3(LPIX / 64, NB), 256, 0, stream>>>(x, off, msk, Bws);
    k_gemm<<<dim3(LPIX / 128, COUT / 128, NB), 256, 0, stream>>>(Aws, Bws, out);
}
